// SpatialAttentionBlock_65816078844082
// MI455X (gfx1250) — compile-verified, matrix operand order checked by reading, not run on hardware
//
#include <hip/hip_runtime.h>
#include <hip/hip_bf16.h>
#include <math.h>

typedef __attribute__((ext_vector_type(16))) _Float16 v16h;
typedef __attribute__((ext_vector_type(8)))  float    v8f;
typedef __attribute__((ext_vector_type(4)))  unsigned u32x4;

#define TOK   25088      // B*C*H*W tokens
#define E_    768
#define NH_   12
#define HD_   64
#define WS_   14
#define NWT   196        // tokens per window
#define HEADS 1536       // 128 windows * 12 heads
#define MLP_  3072
#define SCALE 0.125f     // HD^-0.5

union FragU { v16h v; u32x4 u[2]; };

// A/B fragment for v_wmma_f32_16x16x32_f16.
// p = row base + k0 + half*8 (halves). Lanes 0-15: rows 0-15, K pairs
// [half*8 .. half*8+7] in VGPR0-3 and [16+half*8 .. +7] in VGPR4-7.
__device__ inline v16h ld_frag_h(const _Float16* p) {
  FragU f;
  f.u[0] = *(const u32x4*)p;
  f.u[1] = *(const u32x4*)(p + 16);
  return f.v;
}

// Build an f16 A-fragment from an fp32 row in LDS (softmax probabilities).
__device__ inline v16h frag_from_f32(const float* p, int half) {
  v16h a;
#pragma unroll
  for (int i = 0; i < 8; ++i) {
    int kb = (i < 4) ? (half * 8 + 2 * i) : (16 + half * 8 + 2 * (i - 4));
    a[2 * i]     = (_Float16)p[kb];
    a[2 * i + 1] = (_Float16)p[kb + 1];
  }
  return a;
}

__device__ inline v8f wmma_f16(v16h a, v16h b, v8f c) {
  return __builtin_amdgcn_wmma_f32_16x16x32_f16(false, a, false, b, (short)0, c,
                                                false, false);
}

// ---------------- weight convert + transpose: dst[n*K+k] = (f16)src[k*N+n]
__global__ void k_cvtT(const float* __restrict__ src, _Float16* __restrict__ dst,
                       int K, int N) {
  size_t i = (size_t)blockIdx.x * 256 + threadIdx.x;
  if (i < (size_t)K * N) {
    int k = (int)(i / N), n = (int)(i % N);
    dst[(size_t)n * K + k] = (_Float16)src[i];
  }
}

// ---------------- layernorm (optionally window-partition remap on input)
__global__ void __launch_bounds__(256) k_ln(const float* __restrict__ src,
                                            const float* __restrict__ w,
                                            const float* __restrict__ b,
                                            _Float16* __restrict__ out,
                                            int remap) {
  __shared__ float red[256];
  int t = blockIdx.x, tid = threadIdx.x;
  const float* row;
  if (remap) {                       // out token t is in window order
    int win = t / NWT, i = t % NWT;
    int img = win >> 4, wh = (win & 15) >> 2, ww = win & 3;
    int h = wh * WS_ + i / WS_, wc = ww * WS_ + i % WS_;
    row = src + ((size_t)(img * 56 + h) * 56 + wc) * E_;
  } else {
    row = src + (size_t)t * E_;
  }
  float s = 0.f;
  for (int e = tid; e < E_; e += 256) s += row[e];
  red[tid] = s; __syncthreads();
  for (int o = 128; o > 0; o >>= 1) { if (tid < o) red[tid] += red[tid + o]; __syncthreads(); }
  float mu = red[0] * (1.f / E_);
  __syncthreads();
  float s2 = 0.f;
  for (int e = tid; e < E_; e += 256) { float d = row[e] - mu; s2 += d * d; }
  red[tid] = s2; __syncthreads();
  for (int o = 128; o > 0; o >>= 1) { if (tid < o) red[tid] += red[tid + o]; __syncthreads(); }
  float rstd = rsqrtf(red[0] * (1.f / E_) + 1e-5f);
  _Float16* orow = out + (size_t)t * E_;
  for (int e = tid; e < E_; e += 256)
    orow[e] = (_Float16)((row[e] - mu) * rstd * w[e] + b[e]);
}

// ---------------- generic WMMA GEMM: C[M,N] = A[M,K] * Bt[N,K]^T + bias
// Each wave computes a 32x64 output block (2 M-tiles x 4 N-tiles):
// 6 fragment loads feed 8 WMMAs per k-step.
constexpr int EPI_QKV = 0, EPI_GELU = 1, EPI_PROJ = 2, EPI_MLP2 = 3;

template <int EPI>
__global__ void __launch_bounds__(256) k_gemm(const _Float16* __restrict__ A,
                                              const _Float16* __restrict__ Bt,
                                              const float* __restrict__ bias,
                                              int K,
                                              float* __restrict__ outF,
                                              _Float16* __restrict__ outH,
                                              const float* __restrict__ resid) {
  int wave = threadIdx.x >> 5, lane = threadIdx.x & 31;
  int lrow = lane & 15, half = lane >> 4;
  int mTile0 = blockIdx.x * 16 + wave * 2;   // 8 waves -> 16 M tiles per block
  int ntBase = blockIdx.y * 4;
  const _Float16* arow0 = A + (size_t)(mTile0 * 16 + lrow) * K;
  const _Float16* arow1 = arow0 + (size_t)16 * K;
  const _Float16* brow  = Bt + (size_t)(ntBase * 16 + lrow) * K;
  v8f acc[2][4] = {};
  for (int k0 = 0; k0 < K; k0 += 32) {
    int off = k0 + half * 8;
    v16h a0 = ld_frag_h(arow0 + off);
    v16h a1 = ld_frag_h(arow1 + off);
    v16h b0 = ld_frag_h(brow + off);
    v16h b1 = ld_frag_h(brow + (size_t)16 * K + off);
    v16h b2 = ld_frag_h(brow + (size_t)32 * K + off);
    v16h b3 = ld_frag_h(brow + (size_t)48 * K + off);
    __builtin_prefetch(arow0 + k0 + 128, 0, 1);
    acc[0][0] = wmma_f16(a0, b0, acc[0][0]);
    acc[0][1] = wmma_f16(a0, b1, acc[0][1]);
    acc[0][2] = wmma_f16(a0, b2, acc[0][2]);
    acc[0][3] = wmma_f16(a0, b3, acc[0][3]);
    acc[1][0] = wmma_f16(a1, b0, acc[1][0]);
    acc[1][1] = wmma_f16(a1, b1, acc[1][1]);
    acc[1][2] = wmma_f16(a1, b2, acc[1][2]);
    acc[1][3] = wmma_f16(a1, b3, acc[1][3]);
  }
  // Epilogue: row-outer so per-row index math is hoisted out of the j loop.
#pragma unroll
  for (int mi = 0; mi < 2; ++mi) {
    int mBase = (mTile0 + mi) * 16 + half * 8;   // D: lanes16-31 -> rows+8
#pragma unroll
    for (int r = 0; r < 8; ++r) {
      int mm = mBase + r;
      if constexpr (EPI == EPI_QKV) {
        int win = mm / NWT, i = mm % NWT;
        size_t rowScatter = ((size_t)(win * NH_) * NWT + i) * HD_;
#pragma unroll
        for (int j = 0; j < 4; ++j) {
          int n = (ntBase + j) * 16 + lrow;
          float v = acc[mi][j][r] + bias[n];
          int which = n / E_, rem = n % E_;
          int head = rem >> 6, hd = rem & 63;
          outH[(size_t)which * (HEADS * NWT * HD_) + rowScatter +
               (size_t)head * (NWT * HD_) + hd] = (_Float16)v;
        }
      } else if constexpr (EPI == EPI_GELU) {
        _Float16* orow = outH + (size_t)mm * MLP_;
#pragma unroll
        for (int j = 0; j < 4; ++j) {
          int n = (ntBase + j) * 16 + lrow;
          float v = acc[mi][j][r] + bias[n];
          v = 0.5f * v * (1.f + erff(v * 0.70710678118f));   // exact GELU
          orow[n] = (_Float16)v;
        }
      } else if constexpr (EPI == EPI_PROJ) {
        int win = mm / NWT, i = mm % NWT;                    // window unpartition
        int img = win >> 4, wh = (win & 15) >> 2, ww = win & 3;
        int h = wh * WS_ + i / WS_, wc = ww * WS_ + i % WS_;
        size_t rowFlat = ((size_t)(img * 56 + h) * 56 + wc) * E_;
#pragma unroll
        for (int j = 0; j < 4; ++j) {
          int n = (ntBase + j) * 16 + lrow;
          float v = acc[mi][j][r] + bias[n];
          outF[rowFlat + n] = v + resid[rowFlat + n];        // + shortcut x
        }
      } else {  // EPI_MLP2
        size_t rowFlat = (size_t)mm * E_;
#pragma unroll
        for (int j = 0; j < 4; ++j) {
          int n = (ntBase + j) * 16 + lrow;
          float v = acc[mi][j][r] + bias[n];
          outF[rowFlat + n] = v + resid[rowFlat + n];        // + x1
        }
      }
    }
  }
}

// ---------------- fused attention: one block per (window, head)
// LDS: q[208x64]f16 + k[208x64]f16 + vT[64x224]f16 + relH/relW[208x16]f32
//      + scores[208x224]f32  = 294,912 bytes (< 320KB WGP LDS)
#define ATTN_SMEM ((208 * 64 + 208 * 64 + 64 * 224) * 2 + (208 * 16 * 2) * 4 + 208 * 224 * 4)

__global__ void __launch_bounds__(256) k_attn(const _Float16* __restrict__ qkv,
                                              const float* __restrict__ rph,
                                              const float* __restrict__ rpw,
                                              _Float16* __restrict__ outO) {
  extern __shared__ char smem[];
  _Float16* Sq  = (_Float16*)smem;                 // [208][64]
  _Float16* Sk  = Sq + 208 * 64;                   // [208][64]
  _Float16* SvT = Sk + 208 * 64;                   // [64][224]  (v transposed)
  float* relH = (float*)(SvT + 64 * 224);          // [208][16], zero padded
  float* relW = relH + 208 * 16;                   // [208][16], zero padded
  float* Ssc  = relW + 208 * 16;                   // [208][224]

  int win = blockIdx.x / NH_, head = blockIdx.x % NH_;
  int tid = threadIdx.x;
  size_t baseQ = ((size_t)0 * HEADS + win * NH_ + head) * (NWT * HD_);
  size_t baseK = ((size_t)1 * HEADS + win * NH_ + head) * (NWT * HD_);
  size_t baseV = ((size_t)2 * HEADS + win * NH_ + head) * (NWT * HD_);

  const unsigned* qg = (const unsigned*)(qkv + baseQ);
  const unsigned* kg = (const unsigned*)(qkv + baseK);
  unsigned* Sq32 = (unsigned*)Sq;
  unsigned* Sk32 = (unsigned*)Sk;
  for (int i = tid; i < (NWT * HD_) / 2; i += 256) { Sq32[i] = qg[i]; Sk32[i] = kg[i]; }
  for (int i = (NWT * HD_) / 2 + tid; i < (208 * HD_) / 2; i += 256) { Sq32[i] = 0u; Sk32[i] = 0u; }
  const _Float16* vg = qkv + baseV;
  for (int i = tid; i < NWT * HD_; i += 256) {
    int r = i >> 6, hd = i & 63;
    SvT[hd * 224 + r] = vg[i];
  }
  for (int i = tid; i < HD_ * 28; i += 256) {       // zero padded K cols of vT
    int hd = i / 28, c = NWT + i % 28;
    SvT[hd * 224 + c] = (_Float16)0.f;
  }
  __syncthreads();

  // decomposed rel-pos bias, padded to [208][16] with zeros so the score
  // store below can be unconditional for the full padded tile range.
  for (int p = tid; p < 208 * 16; p += 256) {
    int i = p >> 4, r = p & 15;
    float sh = 0.f, sw = 0.f;
    if (i < NWT && r < WS_) {
      int qh = i / WS_, qw = i % WS_;
      const _Float16* qr = Sq + i * HD_;
      const float* rh = rph + (qh - r + WS_ - 1) * HD_;
      const float* rw = rpw + (qw - r + WS_ - 1) * HD_;
      for (int c = 0; c < HD_; ++c) {
        float qv = (float)qr[c];
        sh += qv * rh[c];
        sw += qv * rw[c];
      }
    }
    relH[p] = sh;
    relW[p] = sw;
  }
  __syncthreads();

  int wave = tid >> 5, lane = tid & 31;
  int lrow = lane & 15, half = lane >> 4;

  // S = (q @ k^T) * scale + rel bias   (13x13 tiles, K=64)
  // Stores are unconditional; padded rows/cols get overwritten by the
  // softmax-phase zeroing before anyone reads them.
  for (int t = wave; t < 13 * 13; t += 8) {
    int mt = t / 13, nt = t % 13;
    const _Float16* arow = Sq + (mt * 16 + lrow) * HD_;
    const _Float16* brow = Sk + (nt * 16 + lrow) * HD_;  // k^T col n == k row n
    v8f acc = {};
    acc = wmma_f16(ld_frag_h(arow + half * 8),      ld_frag_h(brow + half * 8),      acc);
    acc = wmma_f16(ld_frag_h(arow + 32 + half * 8), ld_frag_h(brow + 32 + half * 8), acc);
    int n = nt * 16 + lrow;
    float bh_col = (float)(n / WS_);   // column indices within rel tables
    int ch = n / WS_, cw = n % WS_;
    (void)bh_col;
#pragma unroll
    for (int r = 0; r < 8; ++r) {
      int m = mt * 16 + half * 8 + r;
      Ssc[m * 224 + n] = acc[r] * SCALE + relH[(m << 4) + ch] + relW[(m << 4) + cw];
    }
  }
  __syncthreads();

  // row softmax; zero all K padding so P@V padding contributes nothing
  if (tid < NWT) {
    float* row = Ssc + tid * 224;
    float mx = -3.0e38f;
    for (int n = 0; n < NWT; ++n) mx = fmaxf(mx, row[n]);
    float s = 0.f;
    for (int n = 0; n < NWT; ++n) { float e = __expf(row[n] - mx); row[n] = e; s += e; }
    float inv = 1.f / s;
    for (int n = 0; n < NWT; ++n) row[n] *= inv;
    for (int n = NWT; n < 224; ++n) row[n] = 0.f;
  } else if (tid < 208) {
    float* row = Ssc + tid * 224;
    for (int n = 0; n < 224; ++n) row[n] = 0.f;     // padded query rows -> 0
  }
  __syncthreads();

  // O = P @ V   (13x4 tiles, K padded to 224)
  for (int t = wave; t < 13 * 4; t += 8) {
    int mt = t >> 2, nt = t & 3;
    const float* arow = Ssc + (mt * 16 + lrow) * 224;
    const _Float16* brow = SvT + (nt * 16 + lrow) * 224;
    v8f acc = {};
#pragma unroll
    for (int k0 = 0; k0 < 224; k0 += 32) {
      v16h a = frag_from_f32(arow + k0, half);
      v16h b = ld_frag_h(brow + k0 + half * 8);
      acc = wmma_f16(a, b, acc);
    }
    int hd = nt * 16 + lrow;
    _Float16* obase = outO + ((size_t)(win * NWT + mt * 16 + half * 8)) * E_ +
                      head * HD_ + hd;
    if (mt < 12) {                    // full tile: unconditional stores
#pragma unroll
      for (int r = 0; r < 8; ++r) obase[(size_t)r * E_] = (_Float16)acc[r];
    } else if (half == 0) {           // last tile: only rows 192..195 valid
#pragma unroll
      for (int r = 0; r < 4; ++r) obase[(size_t)r * E_] = (_Float16)acc[r];
    }
  }
}

extern "C" void kernel_launch(void* const* d_in, const int* in_sizes, int n_in,
                              void* d_out, int out_size, void* d_ws, size_t ws_size,
                              hipStream_t stream) {
  const float* x      = (const float*)d_in[0];
  const float* n1w    = (const float*)d_in[1];
  const float* n1b    = (const float*)d_in[2];
  const float* qkv_w  = (const float*)d_in[3];
  const float* qkv_b  = (const float*)d_in[4];
  const float* proj_w = (const float*)d_in[5];
  const float* proj_b = (const float*)d_in[6];
  const float* rph    = (const float*)d_in[7];
  const float* rpw    = (const float*)d_in[8];
  const float* n2w    = (const float*)d_in[9];
  const float* n2b    = (const float*)d_in[10];
  const float* w1     = (const float*)d_in[11];
  const float* b1     = (const float*)d_in[12];
  const float* w2     = (const float*)d_in[13];
  const float* b2     = (const float*)d_in[14];

  // workspace layout (with lifetime-based reuse):
  _Float16* wX    = (_Float16*)d_ws;                       // [25088][768] ln1 out
  _Float16* wQKV  = wX + (size_t)TOK * E_;                 // [3][1536][196][64]
  _Float16* attnO = wQKV + (size_t)3 * HEADS * NWT * HD_;  // [25088][768]
  float*    x1    = (float*)(attnO + (size_t)TOK * E_);    // [25088][768] f32
  _Float16* wTq   = (_Float16*)(x1 + (size_t)TOK * E_);    // transposed f16 weights
  _Float16* wTp   = wTq + (size_t)E_ * 3 * E_;
  _Float16* wTm1  = wTp + (size_t)E_ * E_;
  _Float16* wTm2  = wTm1 + (size_t)E_ * MLP_;
  _Float16* h2     = wX;     // ln2 out reuses wX (dead after qkv GEMM)
  _Float16* hidden = wQKV;   // [25088][3072] == wQKV+attnO region exactly

  k_cvtT<<<(E_ * 3 * E_ + 255) / 256, 256, 0, stream>>>(qkv_w, wTq, E_, 3 * E_);
  k_cvtT<<<(E_ * E_ + 255) / 256, 256, 0, stream>>>(proj_w, wTp, E_, E_);
  k_cvtT<<<(E_ * MLP_ + 255) / 256, 256, 0, stream>>>(w1, wTm1, E_, MLP_);
  k_cvtT<<<(E_ * MLP_ + 255) / 256, 256, 0, stream>>>(w2, wTm2, MLP_, E_);

  k_ln<<<TOK, 256, 0, stream>>>(x, n1w, n1b, wX, 1);

  k_gemm<EPI_QKV><<<dim3(TOK / 256, (3 * E_) / 64), 256, 0, stream>>>(
      wX, wTq, qkv_b, E_, nullptr, wQKV, nullptr);

  hipFuncSetAttribute(reinterpret_cast<const void*>(k_attn),
                      hipFuncAttributeMaxDynamicSharedMemorySize, ATTN_SMEM);
  k_attn<<<HEADS, 256, ATTN_SMEM, stream>>>(wQKV, rph, rpw, attnO);

  k_gemm<EPI_PROJ><<<dim3(TOK / 256, E_ / 64), 256, 0, stream>>>(
      attnO, wTp, proj_b, E_, x1, nullptr, x);

  k_ln<<<TOK, 256, 0, stream>>>(x1, n2w, n2b, h2, 0);

  k_gemm<EPI_GELU><<<dim3(TOK / 256, MLP_ / 64), 256, 0, stream>>>(
      h2, wTm1, b1, E_, nullptr, hidden, nullptr);

  k_gemm<EPI_MLP2><<<dim3(TOK / 256, E_ / 64), 256, 0, stream>>>(
      hidden, wTm2, b2, MLP_, (float*)d_out, nullptr, x1);
}
